// AttentionModule_790273983017
// MI455X (gfx1250) — matrix operand order repaired and checked by reading, not compiled, not run on hardware
//
#include <hip/hip_runtime.h>

typedef _Float16 half_t;
typedef __attribute__((ext_vector_type(16))) _Float16 v16h;
typedef __attribute__((ext_vector_type(8)))  _Float16 v8h;
typedef __attribute__((ext_vector_type(8)))  float    v8f;

#define Bn   8
#define Cc   256
#define Nn   4096
#define CQKd 32
#define OTOT 320            // 32 (q) + 32 (k) + 256 (v) stacked output channels
#define NT16 (Nn / 16)      // 256 n-tiles
#define OT16 (OTOT / 16)    // 20 o-tiles

// ---------------------------------------------------------------------------
// CDNA5 async global->LDS copy (ASYNCcnt path, CDNA5 ISA).
// One instruction moves 16 bytes per active lane (512 B per wave).
// lds_off = LDS byte address (low 32 bits of a __shared__ pointer).
// ---------------------------------------------------------------------------
__device__ __forceinline__ void async_lds_b128(unsigned lds_off, const void* gaddr)
{
    asm volatile("global_load_async_to_lds_b128 %0, %1, off"
                 :: "v"(lds_off), "v"(gaddr)
                 : "memory");
}

__device__ __forceinline__ void wait_async0()
{
    asm volatile("s_wait_asynccnt 0" ::: "memory");
}

// ---------------------------------------------------------------------------
// Kernel 1: fused q/k/v projection.  Out[320 x 4096] = W[320 x 256] @ x[256 x 4096]
// per batch, 16x16 WMMA tiles with K=32 steps (v_wmma_f32_16x16x32_f16).
// Each wave owns ONE x n-tile and TEN stacked o-tiles: the x B operand is
// loaded once per K-step and reused across 10 WMMAs (10 v8f accumulators),
// cutting x re-reads 10x.  Stores q,k as f16 [b][n][32] (channel-contiguous
// -> kernel-2 K tiles are contiguous 2 KB blocks) and v as f16 [b][c][4096]
// (position-contiguous 64 B rows).
// ---------------------------------------------------------------------------
__global__ __launch_bounds__(128)
void proj_kernel(const float* __restrict__ x,
                 const float* __restrict__ Wq, const float* __restrict__ bq,
                 const float* __restrict__ Wk, const float* __restrict__ bk,
                 const float* __restrict__ Wv, const float* __restrict__ bv,
                 half_t* __restrict__ qh, half_t* __restrict__ kh,
                 half_t* __restrict__ vh)
{
    const int lane = threadIdx.x & 31;
    const int wv   = threadIdx.x >> 5;
    const int gw   = blockIdx.x * 4 + wv;            // 4096 waves total
    const int ohalf = gw & 1;                        // which 10 of the 20 o-tiles
    const int nti  = gw >> 1;                        // (b, n-tile) id, 2048 total
    const int b    = nti / NT16;
    const int nt   = nti % NT16;
    const int n0   = nt * 16;
    const bool hi  = lane >= 16;
    const int  lm  = lane & 15;

    // Per-lane A-operand row pointers for the 10 o-tiles this wave owns.
    const float* Wrowp[10];
#pragma unroll
    for (int ot = 0; ot < 10; ++ot) {
        const int o0 = (ohalf * 10 + ot) * 16;
        const float* Wb;
        int ob;
        if (o0 < 32)      { Wb = Wq; ob = o0; }
        else if (o0 < 64) { Wb = Wk; ob = o0 - 32; }
        else              { Wb = Wv; ob = o0 - 64; }
        Wrowp[ot] = Wb + (size_t)(ob + lm) * Cc;
    }

    const float* xcol = x + (size_t)b * Cc * Nn + (n0 + lm);

    v8f acc[10];
#pragma unroll
    for (int ot = 0; ot < 10; ++ot)
#pragma unroll
        for (int r = 0; r < 8; ++r) acc[ot][r] = 0.f;

    for (int ks = 0; ks < 8; ++ks) {                 // K = 256 in steps of 32
        const int k0 = ks * 32;
        // B (this code's contiguous k order): lanes 0-15 K=0..15, lanes 16-31 K=16..31, N = lane%16
        v16h bx;
        const int bc0 = k0 + (hi ? 16 : 0);
#pragma unroll
        for (int j = 0; j < 16; ++j)
            bx[j] = (half_t)xcol[(size_t)(bc0 + j) * Nn];

        // A (contiguous k order, matching B): lanes 0-15 K=0..15, lanes 16-31 K=16..31
        const int aK0 = k0 + (hi ? 16 : 0);
        const int aK1 = k0 + (hi ? 24 : 8);
#pragma unroll
        for (int ot = 0; ot < 10; ++ot) {
            const float* Wrow = Wrowp[ot];
            v16h a;
#pragma unroll
            for (int j = 0; j < 8; ++j) {
                a[j]     = (half_t)Wrow[aK0 + j];
                a[8 + j] = (half_t)Wrow[aK1 + j];
            }
            acc[ot] = __builtin_amdgcn_wmma_f32_16x16x32_f16(false, a, false, bx,
                                                             (short)0, acc[ot],
                                                             false, false);
        }
    }

    // Store: C/D layout -> element (o = o0 + r + (hi?8:0), n = n0 + lane%16)
    const int n = n0 + lm;
#pragma unroll
    for (int ot = 0; ot < 10; ++ot) {
        const int o0 = (ohalf * 10 + ot) * 16;
#pragma unroll
        for (int r = 0; r < 8; ++r) {
            const int o = o0 + r + (hi ? 8 : 0);
            float bias;
            if (o < 32)      bias = bq[o];
            else if (o < 64) bias = bk[o - 32];
            else             bias = bv[o - 64];
            const float val = acc[ot][r] + bias;
            if (o < 32)
                qh[((size_t)b * Nn + n) * CQKd + o] = (half_t)val;
            else if (o < 64)
                kh[((size_t)b * Nn + n) * CQKd + (o - 32)] = (half_t)val;
            else
                vh[((size_t)b * Cc + (o - 64)) * Nn + n] = (half_t)val;
        }
    }
}

// ---------------------------------------------------------------------------
// Stage one 32-key step's K tile (2 KB, contiguous) and V tile (16 KB, 64 B
// rows) into LDS with async b128 copies.  256 threads: every thread issues 4
// V chunks; threads 0..127 (waves 0-3, wave-uniform) issue 1 K chunk.
// ---------------------------------------------------------------------------
__device__ __forceinline__ void stage_tiles(const half_t* __restrict__ kbase,
                                            const half_t* __restrict__ vbase,
                                            int m0, half_t* kS, half_t* vS, int tid)
{
    const unsigned kls = (unsigned)(size_t)kS;
    const unsigned vls = (unsigned)(size_t)vS;
#pragma unroll
    for (int i = 0; i < 4; ++i) {
        const int chunk = tid + 256 * i;             // 1024 chunks = 16 KB V tile
        const int c  = chunk >> 2;                   // channel row (4 chunks/row)
        const int mo = (chunk & 3) * 8;              // half offset within row
        async_lds_b128(vls + (unsigned)chunk * 16,
                       vbase + (size_t)c * Nn + m0 + mo);
    }
    if (tid < 128) {                                 // 128 chunks = 2 KB K tile
        async_lds_b128(kls + (unsigned)tid * 16,
                       kbase + (size_t)m0 * CQKd + tid * 8);
    }
}

// ---------------------------------------------------------------------------
// Kernel 2: fused flash attention + output.  One wave per (batch, 16-query
// tile); 8 waves/block share double-buffered K/V tiles in LDS (36 KB).
// S^T = K_tile(16m x 32c) @ q^T(32c x 16n)  -> queries on lanes, keys on
// VGPRs: row softmax = in-lane reduction + one shfl_xor(16), and exp(S^T) is
// the PV B operand after a single lane-half exchange.
// O(16c x 16n) += V_tile(16c x 32m) @ P^T(32m x 16n), 16 channel tiles.
// ---------------------------------------------------------------------------
__global__ __launch_bounds__(256)
void attn_kernel(const float* __restrict__ x, const float* __restrict__ gamma,
                 const half_t* __restrict__ qh, const half_t* __restrict__ kh,
                 const half_t* __restrict__ vh, float* __restrict__ out)
{
    __shared__ half_t kS[2][32 * CQKd];              // 2 x 2 KB
    __shared__ half_t vS[2][Cc * 32];                // 2 x 16 KB

    const int tid  = threadIdx.x;
    const int lane = tid & 31;
    const int wave = tid >> 5;
    const int gw   = blockIdx.x * 8 + wave;
    const int b    = gw / NT16;                      // same b for all 8 waves
    const int nt   = gw % NT16;
    const int n0   = nt * 16;
    const bool hi  = lane >= 16;
    const int  lm  = lane & 15;
    const float g  = gamma[0];

    // q as B operand (32x16): lanes 0-15 hold c={0..7,16..23}, lanes 16-31 hold c={8..15,24..31}
    const half_t* qrow = qh + ((size_t)b * Nn + (n0 + lm)) * CQKd + (hi ? 8 : 0);
    v16h qB;
    {
        v8h q0 = *(const v8h*)(qrow);
        v8h q1 = *(const v8h*)(qrow + 16);
#pragma unroll
        for (int j = 0; j < 8; ++j) { qB[j] = q0[j]; qB[8 + j] = q1[j]; }
    }

    v8f O[16];
#pragma unroll
    for (int t = 0; t < 16; ++t)
#pragma unroll
        for (int r = 0; r < 8; ++r) O[t][r] = 0.f;

    float mrow = -1e30f;
    float lrow = 0.f;

    const half_t* kbase = kh + (size_t)b * Nn * CQKd;
    const half_t* vbase = vh + (size_t)b * Cc * Nn;
    const v8f zero = {0.f, 0.f, 0.f, 0.f, 0.f, 0.f, 0.f, 0.f};

    stage_tiles(kbase, vbase, 0, kS[0], vS[0], tid);

    for (int mt = 0; mt < Nn / 32; ++mt) {
        const int m0  = mt * 32;
        const int cur = mt & 1;

        wait_async0();               // this wave's copies for buffer `cur` done
        __syncthreads();             // all waves' copies visible; prev buf free

        if (mt + 1 < Nn / 32)        // prefetch next step into the other buffer
            stage_tiles(kbase, vbase, m0 + 32, kS[cur ^ 1], vS[cur ^ 1], tid);

        const half_t* kt = kS[cur];
        const half_t* vt = vS[cur];

        // K tiles as A operands (16-bit 16x32 layout) from LDS
        v16h kA0, kA1;
        {
            const half_t* kr0 = kt + (size_t)lm * CQKd;
            const half_t* kr1 = kt + (size_t)(16 + lm) * CQKd;
            v8h a0 = *(const v8h*)(kr0 + (hi ? 8 : 0));
            v8h a1 = *(const v8h*)(kr0 + (hi ? 24 : 16));
            v8h b0 = *(const v8h*)(kr1 + (hi ? 8 : 0));
            v8h b1 = *(const v8h*)(kr1 + (hi ? 24 : 16));
#pragma unroll
            for (int j = 0; j < 8; ++j) {
                kA0[j] = a0[j]; kA0[8 + j] = a1[j];
                kA1[j] = b0[j]; kA1[8 + j] = b1[j];
            }
        }
        v8f T0 = __builtin_amdgcn_wmma_f32_16x16x32_f16(false, kA0, false, qB,
                                                        (short)0, zero, false, false);
        v8f T1 = __builtin_amdgcn_wmma_f32_16x16x32_f16(false, kA1, false, qB,
                                                        (short)0, zero, false, false);

        // online softmax over the 32 keys of this step (per query = per lane)
        float mx = T0[0];
#pragma unroll
        for (int r = 0; r < 8; ++r) { mx = fmaxf(mx, T0[r]); mx = fmaxf(mx, T1[r]); }
        mx = fmaxf(mx, __shfl_xor(mx, 16, 32));
        const float m_new = fmaxf(mrow, mx);
        const float scale = __expf(mrow - m_new);

        v8f e0, e1;
        float s = 0.f;
#pragma unroll
        for (int r = 0; r < 8; ++r) {
            e0[r] = __expf(T0[r] - m_new);
            e1[r] = __expf(T1[r] - m_new);
            s += e0[r] + e1[r];
        }
        s += __shfl_xor(s, 16, 32);
        lrow = lrow * scale + s;
        mrow = m_new;

#pragma unroll
        for (int t = 0; t < 16; ++t)
#pragma unroll
            for (int r = 0; r < 8; ++r) O[t][r] *= scale;

        // Build P^T as B operand (32x16): lanes 0-15 need m0..m0+15,
        // lanes 16-31 need m0+16..m0+31; one cross-half exchange each.
        v16h pB;
#pragma unroll
        for (int r = 0; r < 8; ++r) {
            const float p0 = __shfl_xor(e0[r], 16, 32);  // partner half of T0
            const float p1 = __shfl_xor(e1[r], 16, 32);  // partner half of T1
            const float lowv  = hi ? p1    : e0[r];      // K_local = r
            const float highv = hi ? e1[r] : p0;         // K_local = 8 + r
            pB[r]     = (half_t)lowv;
            pB[8 + r] = (half_t)highv;
        }

        // PV: 16 channel tiles, A = V(16c x 32m) in 16-bit 16x32 layout, from LDS
#pragma unroll
        for (int ct = 0; ct < 16; ++ct) {
            const half_t* vrow = vt + (size_t)(ct * 16 + lm) * 32;
            v8h a0 = *(const v8h*)(vrow + (hi ? 16 : 0));
            v8h a1 = *(const v8h*)(vrow + (hi ? 24 : 8));
            v16h vA;
#pragma unroll
            for (int j = 0; j < 8; ++j) { vA[j] = a0[j]; vA[8 + j] = a1[j]; }
            O[ct] = __builtin_amdgcn_wmma_f32_16x16x32_f16(false, vA, false, pB,
                                                           (short)0, O[ct], false, false);
        }
    }

    // epilogue: out = gamma * (O / l) + x
    const float inv = 1.0f / lrow;
    const int n = n0 + lm;
#pragma unroll
    for (int ct = 0; ct < 16; ++ct) {
#pragma unroll
        for (int r = 0; r < 8; ++r) {
            const int c = ct * 16 + r + (hi ? 8 : 0);
            const size_t idx = ((size_t)b * Cc + c) * Nn + n;
            out[idx] = g * (O[ct][r] * inv) + x[idx];
        }
    }
}

// ---------------------------------------------------------------------------
extern "C" void kernel_launch(void* const* d_in, const int* in_sizes, int n_in,
                              void* d_out, int out_size, void* d_ws, size_t ws_size,
                              hipStream_t stream)
{
    const float* x     = (const float*)d_in[0];
    const float* Wq    = (const float*)d_in[1];
    const float* bq    = (const float*)d_in[2];
    const float* Wk    = (const float*)d_in[3];
    const float* bk    = (const float*)d_in[4];
    const float* Wv    = (const float*)d_in[5];
    const float* bv    = (const float*)d_in[6];
    const float* gamma = (const float*)d_in[7];
    float* out         = (float*)d_out;

    // workspace: q (2MB) | k (2MB) | v (16MB), all f16
    half_t* qh = (half_t*)d_ws;
    half_t* kh = qh + (size_t)Bn * Nn * CQKd;
    half_t* vh = kh + (size_t)Bn * Nn * CQKd;

    // 4096 waves: one per (batch, n-tile, o-half), 4 waves per 128-thread block
    proj_kernel<<<(Bn * NT16 * 2) / 4, 128, 0, stream>>>(
        x, Wq, bq, Wk, bk, Wv, bv, qh, kh, vh);

    // 8 * 256 = 2048 query tiles, 8 waves per 256-thread block
    attn_kernel<<<(Bn * NT16) / 8, 256, 0, stream>>>(x, gamma, qh, kh, vh, out);
}
